// PerformerAttnBlock_57131654971505
// MI455X (gfx1250) — matrix operand order repaired and checked by reading, not compiled, not run on hardware
//
#include <hip/hip_runtime.h>

// ---------------------------------------------------------------------------
// Performer attention block for gfx1250 (MI455X), bf16 WMMA everywhere.
// B=4, C=IN_DIM=512, T=4096, E=1024, H=16, D=64, F=256
// ---------------------------------------------------------------------------

typedef __bf16 bf16_t;
typedef __bf16 v16bf __attribute__((ext_vector_type(16)));
typedef __bf16 v8bf  __attribute__((ext_vector_type(8)));
typedef float  v8f   __attribute__((ext_vector_type(8)));

#define DIM_B 4
#define DIM_T 4096
#define DIM_C 512
#define DIM_E 1024
#define DIM_H 16
#define DIM_D 64
#define DIM_F 256
#define BT    (DIM_B * DIM_T)      // 16384
#define BH    (DIM_B * DIM_H)      // 64
#define NCHUNK 4
#define TCHUNK (DIM_T / NCHUNK)    // 1024

__device__ __forceinline__ v8f vzero8() {
  v8f z;
#pragma unroll
  for (int i = 0; i < 8; ++i) z[i] = 0.0f;
  return z;
}

// A-fragment (16x32 bf16, MxK) per ISA 7.12.2: lane = half*16 + m;
// e=0..7 -> k = kbase + 8*half + e ; e=8..15 -> k = kbase + 16 + 8*half + (e-8)
// -> two contiguous 16-byte LDS chunks from row m.
__device__ __forceinline__ v16bf load_frag_a(const bf16_t* row, int half, int kbase) {
  v8bf lo = *(const v8bf*)(row + kbase + 8 * half);
  v8bf hi = *(const v8bf*)(row + kbase + 16 + 8 * half);
  v16bf a;
#pragma unroll
  for (int e = 0; e < 8; ++e) { a[e] = lo[e]; a[e + 8] = hi[e]; }
  return a;
}

// B-fragment (32x16 bf16, KxN) from LDS staged column-major Bc[n][k]:
// lane = half*16 + n; e=0..7 -> k = kbase + 8*half + e ; e=8..15 -> k = kbase + 16 + 8*half + (e-8).
__device__ __forceinline__ v16bf load_frag_b(const bf16_t* row, int half, int kbase) {
  v8bf lo = *(const v8bf*)(row + kbase + 8 * half);
  v8bf hi = *(const v8bf*)(row + kbase + 16 + 8 * half);
  v16bf b;
#pragma unroll
  for (int e = 0; e < 8; ++e) { b[e] = lo[e]; b[e + 8] = hi[e]; }
  return b;
}

__device__ __forceinline__ v8f wmma_bf16(v16bf a, v16bf b, v8f c) {
  return __builtin_amdgcn_wmma_f32_16x16x32_bf16(false, a, false, b, (short)0, c,
                                                 false, false);
}

// ---------------------------------------------------------------------------
// pack kernels (tiled LDS transposes, coalesced on both global sides)
// ---------------------------------------------------------------------------
// xb(b,t,c) = bf16(x(b,c,t));  grid (C/32, T/32, B), 256 threads
__global__ __launch_bounds__(256) void pack_x_kernel(const float* __restrict__ x,
                                                     bf16_t* __restrict__ xb) {
  __shared__ float tile[32][33];
  const int c0 = blockIdx.x * 32, t0 = blockIdx.y * 32, b = blockIdx.z;
  const int tx = threadIdx.x & 31, ty4 = threadIdx.x >> 5;
#pragma unroll
  for (int i = 0; i < 4; ++i) {
    int ty = ty4 * 4 + i;
    tile[ty][tx] = x[((size_t)b * DIM_C + c0 + ty) * DIM_T + t0 + tx];
  }
  __syncthreads();
#pragma unroll
  for (int i = 0; i < 4; ++i) {
    int ty = ty4 * 4 + i;
    xb[((size_t)b * DIM_T + t0 + ty) * DIM_C + c0 + tx] = (bf16_t)tile[tx][ty];
  }
}

// Wt(n,k) = bf16(W(k,n));  grid (K/32, N/32)
__global__ __launch_bounds__(256) void pack_wT_kernel(const float* __restrict__ W,
                                                      bf16_t* __restrict__ Wt,
                                                      int Kdim, int Ndim) {
  __shared__ float tile[32][33];
  const int k0 = blockIdx.x * 32, n0 = blockIdx.y * 32;
  const int tx = threadIdx.x & 31, ty4 = threadIdx.x >> 5;
#pragma unroll
  for (int i = 0; i < 4; ++i) {
    int ty = ty4 * 4 + i;
    tile[ty][tx] = W[(size_t)(k0 + ty) * Ndim + n0 + tx];
  }
  __syncthreads();
#pragma unroll
  for (int i = 0; i < 4; ++i) {
    int ty = ty4 * 4 + i;
    Wt[(size_t)(n0 + ty) * Kdim + k0 + tx] = (bf16_t)tile[tx][ty];
  }
}

// ---------------------------------------------------------------------------
// Generic bf16 WMMA GEMM: C(MxN) = A(MxK) @ B(KxN), A row-major (M,K),
// B given PRE-TRANSPOSED as Bt (N,K) row-major.  Block tile 128x128, K-step 32,
// 8 waves (4x2), wave tile 32x64 (8 accums), double-buffered LDS.
// epi: 0 = store f32, 1 = store bf16.
// ---------------------------------------------------------------------------
#define GM_BM 128
#define GM_BN 128
#define GM_BK 32

__global__ __launch_bounds__(256) void gemm_bf16_kernel(
    const bf16_t* __restrict__ A, int lda, const bf16_t* __restrict__ Bt,
    void* __restrict__ Cv, int ldc, int M, int N, int K, int epi) {
  __shared__ __align__(16) bf16_t sA[2][GM_BM][GM_BK];  // [m][k]  2x8KB
  __shared__ __align__(16) bf16_t sB[2][GM_BN][GM_BK];  // Bc [n][k] 2x8KB

  const int tid = threadIdx.x;
  const int lane = tid & 31, w = tid >> 5;
  const int half = lane >> 4, ln = lane & 15;
  const int wm = w >> 1, wn = w & 1;  // 4x2 waves, wave tile 32x64
  const int m0 = blockIdx.y * GM_BM;
  const int n0 = blockIdx.x * GM_BN;

  // staging geometry: 512 chunks of 8 bf16 (16B) per tile, 2 per thread
  const int c0i = tid, c1i = tid + 256;
  const int r0 = c0i >> 2, q0 = (c0i & 3) * 8;
  const int r1 = c1i >> 2, q1 = (c1i & 3) * 8;
  const bf16_t* pA0 = A + (size_t)(m0 + r0) * lda + q0;
  const bf16_t* pA1 = A + (size_t)(m0 + r1) * lda + q1;
  const bf16_t* pB0 = Bt + (size_t)(n0 + r0) * K + q0;
  const bf16_t* pB1 = Bt + (size_t)(n0 + r1) * K + q1;

  v8f acc[2][4];
#pragma unroll
  for (int i = 0; i < 2; ++i)
#pragma unroll
    for (int j = 0; j < 4; ++j) acc[i][j] = vzero8();

  uint4 ra0, ra1, rb0, rb1;
  // prologue: stage tile 0
  ra0 = *(const uint4*)pA0; pA0 += GM_BK;
  ra1 = *(const uint4*)pA1; pA1 += GM_BK;
  rb0 = *(const uint4*)pB0; pB0 += GM_BK;
  rb1 = *(const uint4*)pB1; pB1 += GM_BK;
  *(uint4*)&sA[0][r0][q0] = ra0;
  *(uint4*)&sA[0][r1][q1] = ra1;
  *(uint4*)&sB[0][r0][q0] = rb0;
  *(uint4*)&sB[0][r1][q1] = rb1;
  __syncthreads();

  const int NT = K / GM_BK;
  for (int kt = 0; kt < NT; ++kt) {
    const int buf = kt & 1;
    const bool more = (kt + 1 < NT);
    if (more) {  // issue next-tile global loads before the matrix math
      ra0 = *(const uint4*)pA0; pA0 += GM_BK;
      ra1 = *(const uint4*)pA1; pA1 += GM_BK;
      rb0 = *(const uint4*)pB0; pB0 += GM_BK;
      rb1 = *(const uint4*)pB1; pB1 += GM_BK;
      if (kt + 2 < NT) __builtin_prefetch(pA0, 0, 0);
    }
    v16bf af[2], bfr[4];
#pragma unroll
    for (int i = 0; i < 2; ++i)
      af[i] = load_frag_a(&sA[buf][wm * 32 + 16 * i + ln][0], half, 0);
#pragma unroll
    for (int j = 0; j < 4; ++j)
      bfr[j] = load_frag_b(&sB[buf][wn * 64 + 16 * j + ln][0], half, 0);
#pragma unroll
    for (int i = 0; i < 2; ++i)
#pragma unroll
      for (int j = 0; j < 4; ++j) acc[i][j] = wmma_bf16(af[i], bfr[j], acc[i][j]);
    if (more) {
      *(uint4*)&sA[buf ^ 1][r0][q0] = ra0;
      *(uint4*)&sA[buf ^ 1][r1][q1] = ra1;
      *(uint4*)&sB[buf ^ 1][r0][q0] = rb0;
      *(uint4*)&sB[buf ^ 1][r1][q1] = rb1;
      __syncthreads();
    }
  }

#pragma unroll
  for (int i = 0; i < 2; ++i)
#pragma unroll
    for (int j = 0; j < 4; ++j)
#pragma unroll
      for (int r = 0; r < 8; ++r) {
        int gm = m0 + wm * 32 + 16 * i + r + 8 * half;
        int gn = n0 + wn * 64 + 16 * j + ln;
        float v = acc[i][j][r];
        if (epi == 0)
          ((float*)Cv)[(size_t)gm * ldc + gn] = v;
        else
          ((bf16_t*)Cv)[(size_t)gm * ldc + gn] = (bf16_t)v;
      }
}

// ---------------------------------------------------------------------------
// phi_kv: per (chunk, b*H+h): Kp = exp(K_h @ projT)+eps computed on the fly,
// KVpart += Kp^T @ V_h, KSpart = column-sums(Kp) (deterministic).
// ---------------------------------------------------------------------------
__global__ __launch_bounds__(256) void phi_kv_kernel(
    const bf16_t* __restrict__ Kb, const bf16_t* __restrict__ Vb,
    const float* __restrict__ proj,  // (F, D) f32 row-major == Bc[f][d]
    float* __restrict__ KVpart,      // (NCHUNK, BH, F, D)
    float* __restrict__ KSpart) {    // (NCHUNK, BH, F)
  __shared__ __align__(16) bf16_t sProj[DIM_F][DIM_D];  // Bc [f][d]  32 KB
  __shared__ __align__(16) bf16_t sK[32][DIM_D];        // A  [t][d]   4 KB
  __shared__ __align__(16) bf16_t sKp[DIM_F][32];       // A  [f][t]  16 KB
  __shared__ __align__(16) bf16_t sV[DIM_D][32];        // Bc [d][t]   4 KB

  const int tid = threadIdx.x;
  const int lane = tid & 31, w = tid >> 5;
  const int half = lane >> 4, ln = lane & 15;
  const int chunk = blockIdx.x;
  const int bh = blockIdx.y;
  const int b = bh >> 4, h = bh & 15;
  const size_t tbase = (size_t)b * DIM_T + (size_t)chunk * TCHUNK;

  // stage proj -> bf16 (contiguous, float4 loads)
#pragma unroll
  for (int i = 0; i < 16; ++i) {
    int c = tid + i * 256;  // 4096 float4 chunks
    float4 f = ((const float4*)proj)[c];
    bf16_t tmp[4] = {(bf16_t)f.x, (bf16_t)f.y, (bf16_t)f.z, (bf16_t)f.w};
    *(uint2*)(((bf16_t*)sProj) + (size_t)c * 4) = *(const uint2*)tmp;
  }

  v8f kv[2][4];
#pragma unroll
  for (int i = 0; i < 2; ++i)
#pragma unroll
    for (int j = 0; j < 4; ++j) kv[i][j] = vzero8();
  float ksacc[2] = {0.0f, 0.0f};

  const int tl = tid >> 3;          // 0..31  (row within 32-t tile)
  const int dp8 = (tid & 7) * 8;    // 0..56  (8-elem column chunk)
  __syncthreads();

  for (int ts = 0; ts < TCHUNK; ts += 32) {
    // stage K tile (b128 both sides) and V tile (b128 load, transposed store)
    v8bf kvec = *(const v8bf*)&Kb[(tbase + ts + tl) * DIM_E + h * DIM_D + dp8];
    v8bf vvec = *(const v8bf*)&Vb[(tbase + ts + tl) * DIM_E + h * DIM_D + dp8];
    *(v8bf*)&sK[tl][dp8] = kvec;
#pragma unroll
    for (int e = 0; e < 8; ++e) sV[dp8 + e][tl] = vvec[e];
    __syncthreads();

    // S(32x256) = sK(32x64) @ proj(64x256); wave w owns n-range [w*32, +32)
    v8f s[2][2];
#pragma unroll
    for (int i = 0; i < 2; ++i)
#pragma unroll
      for (int j = 0; j < 2; ++j) s[i][j] = vzero8();
#pragma unroll
    for (int ks = 0; ks < 2; ++ks) {
      v16bf a0 = load_frag_a(&sK[ln][0], half, 32 * ks);
      v16bf a1 = load_frag_a(&sK[16 + ln][0], half, 32 * ks);
      v16bf b0 = load_frag_b(&sProj[w * 32 + ln][0], half, 32 * ks);
      v16bf b1 = load_frag_b(&sProj[w * 32 + 16 + ln][0], half, 32 * ks);
      s[0][0] = wmma_bf16(a0, b0, s[0][0]);
      s[0][1] = wmma_bf16(a0, b1, s[0][1]);
      s[1][0] = wmma_bf16(a1, b0, s[1][0]);
      s[1][1] = wmma_bf16(a1, b1, s[1][1]);
    }
    // exp + eps -> sKp[f][t]; per-lane deterministic K_sum accumulation
#pragma unroll
    for (int j = 0; j < 2; ++j) {
      int f = w * 32 + 16 * j + ln;
      float colsum = 0.0f;
#pragma unroll
      for (int i = 0; i < 2; ++i)
#pragma unroll
        for (int r = 0; r < 8; ++r) {
          int t = 16 * i + r + 8 * half;
          float e = __expf(s[i][j][r]) + 1e-6f;
          sKp[f][t] = (bf16_t)e;
          colsum += e;
        }
      ksacc[j] += colsum;
    }
    __syncthreads();

    // KV += Kp^T(256x32) @ V(32x64); wave w owns f-range [w*32, +32)
    v16bf afr[2], bfr[4];
#pragma unroll
    for (int i = 0; i < 2; ++i)
      afr[i] = load_frag_a(&sKp[w * 32 + 16 * i + ln][0], half, 0);
#pragma unroll
    for (int j = 0; j < 4; ++j)
      bfr[j] = load_frag_b(&sV[16 * j + ln][0], half, 0);
#pragma unroll
    for (int i = 0; i < 2; ++i)
#pragma unroll
      for (int j = 0; j < 4; ++j) kv[i][j] = wmma_bf16(afr[i], bfr[j], kv[i][j]);
    __syncthreads();
  }

  float* kvout = KVpart + ((size_t)chunk * BH + bh) * DIM_F * DIM_D;
#pragma unroll
  for (int i = 0; i < 2; ++i)
#pragma unroll
    for (int j = 0; j < 4; ++j)
#pragma unroll
      for (int r = 0; r < 8; ++r) {
        int f = w * 32 + 16 * i + r + 8 * half;
        int d = 16 * j + ln;
        kvout[(size_t)f * DIM_D + d] = kv[i][j][r];
      }
  // lane L and L^16 share the same f -> combine halves, deterministic
  float* ksout = KSpart + ((size_t)chunk * BH + bh) * DIM_F;
#pragma unroll
  for (int j = 0; j < 2; ++j) {
    float tot = ksacc[j] + __shfl_xor(ksacc[j], 16, 32);
    if (half == 0) ksout[w * 32 + 16 * j + ln] = tot;
  }
}

// ---------------------------------------------------------------------------
// reduce NCHUNK partial KV / K_sum slices
// ---------------------------------------------------------------------------
__global__ __launch_bounds__(256) void kv_reduce_kernel(
    const float* __restrict__ KVpart, const float* __restrict__ KSpart,
    float* __restrict__ KV, float* __restrict__ KS) {
  const size_t KVN = (size_t)BH * DIM_F * DIM_D;  // 1048576
  const size_t KSN = (size_t)BH * DIM_F;          // 16384
  size_t idx = (size_t)blockIdx.x * 256 + threadIdx.x;
  if (idx < KVN) {
    float s = 0.0f;
#pragma unroll
    for (int c = 0; c < NCHUNK; ++c) s += KVpart[(size_t)c * KVN + idx];
    KV[idx] = s;
  } else if (idx < KVN + KSN) {
    size_t i2 = idx - KVN;
    float s = 0.0f;
#pragma unroll
    for (int c = 0; c < NCHUNK; ++c) s += KSpart[(size_t)c * KSN + i2];
    KS[i2] = s;
  }
}

// ---------------------------------------------------------------------------
// phi_attn: per (t-tile of 64, b*H+h): Qp = exp(Q_h @ projT)+eps,
// denom = max(Qp @ K_sum, 1e-6), attn = (Qp @ KV) / denom -> bf16.
// ---------------------------------------------------------------------------
__global__ __launch_bounds__(256) void phi_attn_kernel(
    const bf16_t* __restrict__ Qb, const float* __restrict__ proj,
    const float* __restrict__ KV, const float* __restrict__ KS,
    bf16_t* __restrict__ attnb) {
  __shared__ __align__(16) bf16_t sProj[DIM_F][DIM_D];  // Bc [f][d]  32 KB
  __shared__ __align__(16) bf16_t sQ[64][DIM_D];        // A  [t][d]   8 KB
  __shared__ __align__(16) bf16_t sQp[64][DIM_F];       // A  [t][f]  32 KB
  __shared__ __align__(16) bf16_t sKV[DIM_D][DIM_F];    // Bc [d][f]  32 KB
  __shared__ float sKS[DIM_F];
  __shared__ float sDen[64];

  const int tid = threadIdx.x;
  const int lane = tid & 31, w = tid >> 5;
  const int half = lane >> 4, ln = lane & 15;
  const int tt = blockIdx.x;
  const int bh = blockIdx.y;
  const int b = bh >> 4, h = bh & 15;
  const size_t tbase = (size_t)b * DIM_T + (size_t)tt * 64;

#pragma unroll
  for (int i = 0; i < 16; ++i) {
    int c = tid + i * 256;
    float4 f = ((const float4*)proj)[c];
    bf16_t tmp[4] = {(bf16_t)f.x, (bf16_t)f.y, (bf16_t)f.z, (bf16_t)f.w};
    *(uint2*)(((bf16_t*)sProj) + (size_t)c * 4) = *(const uint2*)tmp;
  }
  // KV (f32, [f][d]) -> bf16 Bc [d][f] (float4 loads, transposed b16 stores)
  const float* kvsrc = KV + (size_t)bh * DIM_F * DIM_D;
#pragma unroll
  for (int i = 0; i < 16; ++i) {
    int c = tid + i * 256;  // 4096 float4 chunks
    int f = c >> 4, d4 = (c & 15) * 4;
    float4 g = ((const float4*)kvsrc)[c];
    sKV[d4 + 0][f] = (bf16_t)g.x;
    sKV[d4 + 1][f] = (bf16_t)g.y;
    sKV[d4 + 2][f] = (bf16_t)g.z;
    sKV[d4 + 3][f] = (bf16_t)g.w;
  }
  sKS[tid] = KS[(size_t)bh * DIM_F + tid];
#pragma unroll
  for (int i = 0; i < 2; ++i) {
    int c = tid + i * 256;  // 512 chunks of 8 bf16
    int tl = c >> 3, dp8 = (c & 7) * 8;
    *(v8bf*)&sQ[tl][dp8] =
        *(const v8bf*)&Qb[(tbase + tl) * DIM_E + h * DIM_D + dp8];
  }
  __syncthreads();

  // S(64x256) = sQ @ proj; waves 2x4: wm over m 32, wn over n 64
  {
    const int wm = w >> 2, wn = w & 3;
    v8f s[2][4];
#pragma unroll
    for (int i = 0; i < 2; ++i)
#pragma unroll
      for (int j = 0; j < 4; ++j) s[i][j] = vzero8();
#pragma unroll
    for (int ks = 0; ks < 2; ++ks) {
      v16bf a[2];
#pragma unroll
      for (int i = 0; i < 2; ++i)
        a[i] = load_frag_a(&sQ[wm * 32 + 16 * i + ln][0], half, 32 * ks);
      v16bf bb[4];
#pragma unroll
      for (int j = 0; j < 4; ++j)
        bb[j] = load_frag_b(&sProj[wn * 64 + 16 * j + ln][0], half, 32 * ks);
#pragma unroll
      for (int i = 0; i < 2; ++i)
#pragma unroll
        for (int j = 0; j < 4; ++j) s[i][j] = wmma_bf16(a[i], bb[j], s[i][j]);
    }
#pragma unroll
    for (int i = 0; i < 2; ++i)
#pragma unroll
      for (int j = 0; j < 4; ++j)
#pragma unroll
        for (int r = 0; r < 8; ++r) {
          int t = wm * 32 + 16 * i + r + 8 * half;
          int f = wn * 64 + 16 * j + ln;
          sQp[t][f] = (bf16_t)(__expf(s[i][j][r]) + 1e-6f);
        }
  }
  __syncthreads();

  // denom per t-row
  if (tid < 64) {
    float acc = 0.0f;
#pragma unroll 8
    for (int f = 0; f < DIM_F; ++f) acc += (float)sQp[tid][f] * sKS[f];
    sDen[tid] = fmaxf(acc, 1e-6f);
  }
  __syncthreads();

  // numer(64x64) = Qp(64x256) @ KV(256x64); waves 4x2: wm2 over m 16, wn2 over n 32
  {
    const int wm2 = w >> 1, wn2 = w & 1;
    v8f n[2];
    n[0] = vzero8();
    n[1] = vzero8();
#pragma unroll
    for (int ks = 0; ks < 8; ++ks) {
      v16bf a = load_frag_a(&sQp[wm2 * 16 + ln][0], half, 32 * ks);
      v16bf b0 = load_frag_b(&sKV[wn2 * 32 + ln][0], half, 32 * ks);
      v16bf b1 = load_frag_b(&sKV[wn2 * 32 + 16 + ln][0], half, 32 * ks);
      n[0] = wmma_bf16(a, b0, n[0]);
      n[1] = wmma_bf16(a, b1, n[1]);
    }
#pragma unroll
    for (int j = 0; j < 2; ++j)
#pragma unroll
      for (int r = 0; r < 8; ++r) {
        int t = wm2 * 16 + r + 8 * half;
        int d = wn2 * 32 + 16 * j + ln;
        float v = n[j][r] / sDen[t];
        attnb[(tbase + t) * DIM_E + h * DIM_D + d] = (bf16_t)v;
      }
  }
}

// ---------------------------------------------------------------------------
// output assembly: x passthrough (contiguous copy) + out2 transpose (tiled)
// ---------------------------------------------------------------------------
__global__ __launch_bounds__(256) void copy_x_kernel(const float* __restrict__ x,
                                                     float* __restrict__ out) {
  size_t idx = (size_t)blockIdx.x * 256 + threadIdx.x;  // B*C*T
  const size_t per_b = (size_t)DIM_C * DIM_T;
  size_t b = idx / per_b, r = idx % per_b;
  out[b * (size_t)(DIM_C + DIM_E) * DIM_T + r] = x[idx];
}

// out(b, 512+c, t) = out2(b*T+t, c);  grid (E/32, T/32, B)
__global__ __launch_bounds__(256) void out2_transpose_kernel(
    const float* __restrict__ out2, float* __restrict__ out) {
  __shared__ float tile[32][33];
  const int c0 = blockIdx.x * 32, t0 = blockIdx.y * 32, b = blockIdx.z;
  const int tx = threadIdx.x & 31, ty4 = threadIdx.x >> 5;
#pragma unroll
  for (int i = 0; i < 4; ++i) {
    int ty = ty4 * 4 + i;
    tile[ty][tx] = out2[((size_t)b * DIM_T + t0 + ty) * DIM_E + c0 + tx];
  }
  __syncthreads();
#pragma unroll
  for (int i = 0; i < 4; ++i) {
    int ty = ty4 * 4 + i;
    out[((size_t)b * (DIM_C + DIM_E) + DIM_C + c0 + ty) * DIM_T + t0 + tx] =
        tile[tx][ty];
  }
}

// ---------------------------------------------------------------------------
// host launcher
// ---------------------------------------------------------------------------
extern "C" void kernel_launch(void* const* d_in, const int* in_sizes, int n_in,
                              void* d_out, int out_size, void* d_ws, size_t ws_size,
                              hipStream_t stream) {
  const float* x    = (const float*)d_in[0];
  const float* Wq   = (const float*)d_in[1];
  const float* Wk   = (const float*)d_in[2];
  const float* Wv   = (const float*)d_in[3];
  const float* Wo   = (const float*)d_in[4];
  const float* proj = (const float*)d_in[5];

  char* ws = (char*)d_ws;
  size_t off = 0;
  auto alloc = [&](size_t bytes) -> size_t {
    size_t r = off;
    off += (bytes + 255) & ~(size_t)255;
    return r;
  };
  size_t o_xb   = alloc((size_t)BT * DIM_C * sizeof(bf16_t));
  size_t o_wq   = alloc((size_t)DIM_C * DIM_E * sizeof(bf16_t));
  size_t o_wk   = alloc((size_t)DIM_C * DIM_E * sizeof(bf16_t));
  size_t o_wv   = alloc((size_t)DIM_C * DIM_E * sizeof(bf16_t));
  size_t o_wo   = alloc((size_t)DIM_E * DIM_E * sizeof(bf16_t));
  size_t o_q    = alloc((size_t)BT * DIM_E * sizeof(bf16_t));  // 32 MB
  size_t o_k    = alloc((size_t)BT * DIM_E * sizeof(bf16_t));  // 32 MB
  size_t o_v    = alloc((size_t)BT * DIM_E * sizeof(bf16_t));
  size_t o_attn = alloc((size_t)BT * DIM_E * sizeof(bf16_t));
  size_t o_kvp  = alloc((size_t)NCHUNK * BH * DIM_F * DIM_D * sizeof(float));
  size_t o_ksp  = alloc((size_t)NCHUNK * BH * DIM_F * sizeof(float));
  size_t o_kv   = alloc((size_t)BH * DIM_F * DIM_D * sizeof(float));
  size_t o_ks   = alloc((size_t)BH * DIM_F * sizeof(float));
  size_t o_out2 = o_q;  // alias: Q+K bf16 region (exactly 64 MB) dead before out2

  bf16_t* xb  = (bf16_t*)(ws + o_xb);
  bf16_t* WqT = (bf16_t*)(ws + o_wq);
  bf16_t* WkT = (bf16_t*)(ws + o_wk);
  bf16_t* WvT = (bf16_t*)(ws + o_wv);
  bf16_t* WoT = (bf16_t*)(ws + o_wo);
  bf16_t* Qb  = (bf16_t*)(ws + o_q);
  bf16_t* Kb  = (bf16_t*)(ws + o_k);
  bf16_t* Vb  = (bf16_t*)(ws + o_v);
  bf16_t* Ab  = (bf16_t*)(ws + o_attn);
  float* KVp  = (float*)(ws + o_kvp);
  float* KSp  = (float*)(ws + o_ksp);
  float* KVf  = (float*)(ws + o_kv);
  float* KSf  = (float*)(ws + o_ks);
  float* out2 = (float*)(ws + o_out2);

  // 1) pack inputs: x transpose->bf16; weights transpose->bf16 (N,K)
  pack_x_kernel<<<dim3(DIM_C / 32, DIM_T / 32, DIM_B), 256, 0, stream>>>(x, xb);
  pack_wT_kernel<<<dim3(DIM_C / 32, DIM_E / 32), 256, 0, stream>>>(Wq, WqT, DIM_C, DIM_E);
  pack_wT_kernel<<<dim3(DIM_C / 32, DIM_E / 32), 256, 0, stream>>>(Wk, WkT, DIM_C, DIM_E);
  pack_wT_kernel<<<dim3(DIM_C / 32, DIM_E / 32), 256, 0, stream>>>(Wv, WvT, DIM_C, DIM_E);
  pack_wT_kernel<<<dim3(DIM_E / 32, DIM_E / 32), 256, 0, stream>>>(Wo, WoT, DIM_E, DIM_E);

  // 2) QKV projections (WMMA GEMMs, bf16 out)
  dim3 gP(DIM_E / GM_BN, BT / GM_BM);
  gemm_bf16_kernel<<<gP, 256, 0, stream>>>(xb, DIM_C, WqT, Qb, DIM_E, BT, DIM_E, DIM_C, 1);
  gemm_bf16_kernel<<<gP, 256, 0, stream>>>(xb, DIM_C, WkT, Kb, DIM_E, BT, DIM_E, DIM_C, 1);
  gemm_bf16_kernel<<<gP, 256, 0, stream>>>(xb, DIM_C, WvT, Vb, DIM_E, BT, DIM_E, DIM_C, 1);

  // 3) fused feature map + KV / K_sum, then chunk reduction
  phi_kv_kernel<<<dim3(NCHUNK, BH), 256, 0, stream>>>(Kb, Vb, proj, KVp, KSp);
  size_t redN = (size_t)BH * DIM_F * DIM_D + (size_t)BH * DIM_F;
  kv_reduce_kernel<<<(unsigned)((redN + 255) / 256), 256, 0, stream>>>(KVp, KSp, KVf, KSf);

  // 4) fused Qp / numer / denom -> attn (bf16)
  phi_attn_kernel<<<dim3(DIM_T / 64, BH), 256, 0, stream>>>(Qb, proj, KVf, KSf, Ab);

  // 5) output projection (WMMA GEMM, f32 out; overwrites dead Q/K region)
  gemm_bf16_kernel<<<gP, 256, 0, stream>>>(Ab, DIM_E, WoT, out2, DIM_E, BT, DIM_E, DIM_E, 0);

  // 6) assemble output (B, C+E, T)
  copy_x_kernel<<<(DIM_B * DIM_C * DIM_T) / 256, 256, 0, stream>>>(x, (float*)d_out);
  out2_transpose_kernel<<<dim3(DIM_E / 32, DIM_T / 32, DIM_B), 256, 0, stream>>>(
      out2, (float*)d_out);
}
